// SingleHeadSelfAttention_65420941852788
// MI455X (gfx1250) — matrix operand order repaired and checked by reading, not compiled, not run on hardware
//
#include <hip/hip_runtime.h>
#include <hip/hip_bf16.h>

// ---------------------------------------------------------------------------
// MI455X (gfx1250, wave32). Heavy GEMMs on v_wmma_f32_16x16x32_bf16.
// Weights staged to LDS (TDM tensor_load_to_lds where available), activations
// carried as bf16 with f32 accumulate.
// ---------------------------------------------------------------------------

typedef __attribute__((ext_vector_type(16))) __bf16 v16bf;
typedef __attribute__((ext_vector_type(8)))  float  v8f;
typedef __attribute__((ext_vector_type(4)))  unsigned int v4u;
typedef __attribute__((ext_vector_type(8)))  int v8i;
typedef __attribute__((ext_vector_type(4)))  int v4i;

constexpr int NB    = 32;      // batch
constexpr int NC    = 256;     // channels
constexpr int CDIM  = 64;      // compressed dim
constexpr int HH    = 60;
constexpr int WW    = 60;
constexpr int HWSZ  = HH * WW;    // 3600
constexpr int NTOK  = 54;         // pooled tokens
constexpr int NTILE = HWSZ / 16;  // 225
constexpr int TPB   = 256;        // 8 waves per block
constexpr int WPB   = 8;

#if defined(__has_builtin)
#if __has_builtin(__builtin_amdgcn_tensor_load_to_lds)
#define HAVE_TDM 1
#else
#define HAVE_TDM 0
#endif
#if __has_builtin(__builtin_amdgcn_s_wait_tensorcnt)
#define HAVE_WTC 1
#else
#define HAVE_WTC 0
#endif
#else
#define HAVE_TDM 0
#define HAVE_WTC 0
#endif

__device__ __forceinline__ __bf16 tobf(float f) { return (__bf16)f; }

__device__ __forceinline__ void lds_fence() {
  asm volatile("s_wait_dscnt 0" ::: "memory");
}

struct B32B { uint4 lo, hi; };
__device__ __forceinline__ v16bf combine16(uint4 lo, uint4 hi) {
  B32B t{lo, hi};
  return __builtin_bit_cast(v16bf, t);
}

__device__ __forceinline__ v8f wmma_bf16(v16bf a, v16bf b, v8f c) {
  return __builtin_amdgcn_wmma_f32_16x16x32_bf16(
      false, a, false, b, (short)0, c, false, false);
}

// ---- fragment loaders ------------------------------------------------------
__device__ __forceinline__ v16bf load_a_bf(const __bf16* base, int ld,
                                           int row0, int col0, int lane) {
  const int m  = row0 + (lane & 15);
  const int k0 = col0 + ((lane >> 4) << 4);
  const __bf16* p = base + (size_t)m * ld + k0;
  uint4 lo = *(const uint4*)p;
  uint4 hi = *(const uint4*)(p + 8);
  return combine16(lo, hi);
}

__device__ __forceinline__ v16bf load_b_nmajor(const __bf16* base, int ld,
                                               int n0, int k0, int lane) {
  const int n  = n0 + (lane & 15);
  const int kk = k0 + ((lane >> 4) << 4);
  const __bf16* p = base + (size_t)n * ld + kk;
  uint4 lo = *(const uint4*)p;
  uint4 hi = *(const uint4*)(p + 8);
  return combine16(lo, hi);
}

__device__ __forceinline__ v16bf load_b_f32(const float* __restrict__ src, int ld,
                                            int k0, int n0, int lane) {
  const int n  = n0 + (lane & 15);
  const int kk = k0 + ((lane >> 4) << 4);
  const float* p = src + (size_t)kk * ld + n;
  v16bf b;
#pragma unroll
  for (int j = 0; j < 16; ++j) b[j] = tobf(p[(size_t)j * ld]);
  return b;
}

union Pk8 { uint4 u; __bf16 h[8]; };

// ---- TDM flat copy (global bf16 -> LDS), D# per CDNA5 ISA ------------------
__device__ __forceinline__ void tdm_copy_to_lds(const void* gsrc, void* ldsDst,
                                                int nElems /*bf16, <=65535*/) {
#if HAVE_TDM
  const unsigned long long ga = (unsigned long long)(uintptr_t)gsrc;
  const unsigned int la = (unsigned int)(uintptr_t)ldsDst;
  v4u g0;
  g0[0] = 1u;                                   // count=1 user descriptor
  g0[1] = la;                                   // lds_addr (bytes)
  g0[2] = (unsigned int)ga;                     // global_addr[31:0]
  g0[3] = ((unsigned int)(ga >> 32) & 0x01FFFFFFu) | 0x80000000u; // [56:32]|type=2
  v8i g1;
  g1[0] = 0x00010000;                           // data_size=1 (2B)
  g1[1] = (int)((nElems & 0xFFFF) << 16);       // tensor_dim0[15:0]
  g1[2] = (int)(((nElems >> 16) & 0xFFFF) | (1u << 16)); // dim0 hi | tensor_dim1=1
  g1[3] = (int)((nElems & 0xFFFF) << 16);       // tile_dim0
  g1[4] = 1;                                    // tile_dim1=1
  g1[5] = nElems;                               // tensor_dim0_stride lo32
  g1[6] = 0;
  g1[7] = 0;
  v4i z4 = {0, 0, 0, 0};
  v8i z8 = {0, 0, 0, 0, 0, 0, 0, 0};
  __builtin_amdgcn_tensor_load_to_lds(g0, g1, z4, z4, z8, 0);
#else
  (void)gsrc; (void)ldsDst; (void)nElems;
#endif
}

__device__ __forceinline__ void tdm_wait0() {
#if HAVE_WTC
  __builtin_amdgcn_s_wait_tensorcnt((short)0);
#else
  asm volatile("s_wait_tensorcnt 0" ::: "memory");
#endif
}

// ---------------------------------------------------------------------------
__global__ __launch_bounds__(256) void convert_weights_kernel(
    const float* __restrict__ w_compress, const float* __restrict__ wq,
    const float* __restrict__ wr,
    __bf16* __restrict__ wc_bf, __bf16* __restrict__ wq_bf,
    __bf16* __restrict__ wr_bf) {
  const int i = blockIdx.x * 256 + threadIdx.x;
  if (i < CDIM * NC) wc_bf[i] = tobf(w_compress[i]);
  if (i < CDIM * CDIM) wq_bf[i] = tobf(wq[i]);
  if (i < NC * CDIM) wr_bf[i] = tobf(wr[i]);
}

// ---------------------------------------------------------------------------
__global__ __launch_bounds__(TPB) void compress_kernel(
    const float* __restrict__ x, const __bf16* __restrict__ wc_bf,
    const float* __restrict__ bias, float* __restrict__ xc,
    __bf16* __restrict__ xcT) {
  __shared__ __attribute__((aligned(16))) __bf16 wlds[CDIM * NC];   // 32KB

#if HAVE_TDM
  if (threadIdx.x == 0) {
    tdm_copy_to_lds(wc_bf, wlds, CDIM * NC);
    tdm_wait0();
  }
#else
  for (int i = threadIdx.x; i < (CDIM * NC) / 8; i += TPB)
    ((uint4*)wlds)[i] = ((const uint4*)wc_bf)[i];
#endif
  __syncthreads();

  const int wave = threadIdx.x >> 5;
  const int lane = threadIdx.x & 31;
  const int tile = blockIdx.x * WPB + wave;
  const int b    = blockIdx.y;
  if (tile >= NTILE) return;
  const int n0 = tile * 16;
  const float* xb   = x   + (size_t)b * NC * HWSZ;
  float*       xcb  = xc  + (size_t)b * CDIM * HWSZ;
  __bf16*      xcTb = xcT + (size_t)b * HWSZ * CDIM;

  v8f acc[4] = {};
#pragma unroll
  for (int kb = 0; kb < NC; kb += 32) {
    v16bf bf = load_b_f32(xb, HWSZ, kb, n0, lane);
#pragma unroll
    for (int mt = 0; mt < 4; ++mt)
      acc[mt] = wmma_bf16(load_a_bf(wlds, NC, mt * 16, kb, lane), bf, acc[mt]);
  }

  const int n  = n0 + (lane & 15);
  const int mh = (lane >> 4) << 3;
#pragma unroll
  for (int mt = 0; mt < 4; ++mt) {
    Pk8 pk;
#pragma unroll
    for (int r = 0; r < 8; ++r) {
      const int m = mt * 16 + mh + r;
      const float v = acc[mt][r] + bias[m];
      xcb[(size_t)m * HWSZ + n] = v;
      pk.h[r] = tobf(v);
    }
    *(uint4*)&xcTb[(size_t)n * CDIM + mt * 16 + mh] = pk.u;
  }
}

// ---------------------------------------------------------------------------
__device__ __forceinline__ void region_of(int m, int& r, int& cell, int& gi, int& base) {
  if (m < 4)       { r = 0; gi = 2; base = 0;  }
  else if (m < 13) { r = 1; gi = 3; base = 4;  }
  else if (m < 29) { r = 2; gi = 4; base = 13; }
  else             { r = 3; gi = 5; base = 29; }
  cell = m - base;
}

__global__ __launch_bounds__(256) void pool_kv_kernel(
    const float* __restrict__ xc,
    const float* __restrict__ w_dsc, const float* __restrict__ b_dsc,
    const float* __restrict__ ln_g,  const float* __restrict__ ln_b,
    const float* __restrict__ wk, const float* __restrict__ bk,
    const float* __restrict__ wv, const float* __restrict__ bv,
    __bf16* __restrict__ Kp, __bf16* __restrict__ Vt) {
  __shared__ float ps[NTOK * CDIM];
  __shared__ float pd[NTOK * CDIM];
  __shared__ float pl[NTOK * CDIM];
  const int b = blockIdx.x;
  const int t = threadIdx.x;
  const float* xcb = xc + (size_t)b * CDIM * HWSZ;

  for (int task = t; task < NTOK * CDIM; task += 256) {
    const int m = task >> 6, ch = task & 63;
    int r, cell, gi, base;
    region_of(m, r, cell, gi, base);
    const int bs = HH / gi;
    const int cy = cell / gi, cx = cell % gi;
    const float* src = xcb + (size_t)ch * HWSZ + (cy * bs) * WW + cx * bs;
    float s = 0.f;
    for (int y = 0; y < bs; ++y)
      for (int xx = 0; xx < bs; ++xx) s += src[y * WW + xx];
    ps[m * CDIM + ch] = s * (1.0f / (bs * bs));
  }
  __syncthreads();

  for (int task = t; task < NTOK * CDIM; task += 256) {
    const int m = task >> 6, ch = task & 63;
    int r, cell, gi, base;
    region_of(m, r, cell, gi, base);
    const int cy = cell / gi, cx = cell % gi;
    const float* kk = w_dsc + (size_t)(r * CDIM + ch) * 9;
    float s = 0.f;
    for (int dy = -1; dy <= 1; ++dy)
      for (int dx = -1; dx <= 1; ++dx) {
        const int yy = cy + dy, xx = cx + dx;
        if (yy >= 0 && yy < gi && xx >= 0 && xx < gi)
          s += kk[(dy + 1) * 3 + (dx + 1)] * ps[(base + yy * gi + xx) * CDIM + ch];
      }
    pd[m * CDIM + ch] = ps[m * CDIM + ch] + s + b_dsc[r * CDIM + ch];
  }
  __syncthreads();

  if (t < NTOK) {
    float mu = 0.f;
    for (int c = 0; c < CDIM; ++c) mu += pd[t * CDIM + c];
    mu *= (1.0f / CDIM);
    float var = 0.f;
    for (int c = 0; c < CDIM; ++c) { const float d = pd[t * CDIM + c] - mu; var += d * d; }
    var *= (1.0f / CDIM);
    const float inv = rsqrtf(var + 1e-5f);
    for (int c = 0; c < CDIM; ++c)
      pl[t * CDIM + c] = (pd[t * CDIM + c] - mu) * inv * ln_g[c] + ln_b[c];
  }
  __syncthreads();

  __bf16* Kb = Kp + (size_t)b * 64 * 64;
  __bf16* Vb = Vt + (size_t)b * 64 * 64;
  for (int task = t; task < 64 * 64; task += 256) {
    const int m = task >> 6, o = task & 63;
    float kv = 0.f, vv = 0.f;
    if (m < NTOK) {
      for (int c = 0; c < CDIM; ++c) {
        const float p = pl[m * CDIM + c];
        kv += p * wk[o * CDIM + c];
        vv += p * wv[o * CDIM + c];
      }
      kv += bk[o];
      vv += bv[o];
    }
    Kb[m * 64 + o] = tobf(kv);
    Vb[o * 64 + m] = tobf(vv);
  }
}

// ---------------------------------------------------------------------------
__global__ __launch_bounds__(TPB) void attn_restore_kernel(
    const __bf16* __restrict__ xcT,
    const __bf16* __restrict__ wq_bf, const float* __restrict__ bq,
    const __bf16* __restrict__ Kp, const __bf16* __restrict__ Vt,
    const __bf16* __restrict__ wr_bf, const float* __restrict__ br,
    float* __restrict__ out) {
  __shared__ __attribute__((aligned(16))) __bf16 wq_l[CDIM * CDIM];
  __shared__ __attribute__((aligned(16))) __bf16 K_l[64 * 64];
  __shared__ __attribute__((aligned(16))) __bf16 V_l[64 * 64];
  __shared__ __attribute__((aligned(16))) __bf16 wr_l[NC * CDIM];
  __shared__ __attribute__((aligned(16))) __bf16 stage[WPB][16 * 64];

  const int b = blockIdx.y;
#if HAVE_TDM
  if (threadIdx.x == 0) tdm_copy_to_lds(wr_bf, wr_l, NC * CDIM);
#else
  for (int i = threadIdx.x; i < (NC * CDIM) / 8; i += TPB)
    ((uint4*)wr_l)[i] = ((const uint4*)wr_bf)[i];
#endif
  {
    const uint4* kg = (const uint4*)(Kp + (size_t)b * 64 * 64);
    const uint4* vg = (const uint4*)(Vt + (size_t)b * 64 * 64);
    const uint4* qg = (const uint4*)wq_bf;
    for (int i = threadIdx.x; i < 512; i += TPB) {
      ((uint4*)wq_l)[i] = qg[i];
      ((uint4*)K_l)[i]  = kg[i];
      ((uint4*)V_l)[i]  = vg[i];
    }
  }
#if HAVE_TDM
  if (threadIdx.x == 0) tdm_wait0();
#endif
  __syncthreads();

  const int wave = threadIdx.x >> 5;
  const int lane = threadIdx.x & 31;
  const int tile = blockIdx.x * WPB + wave;
  if (tile >= NTILE) return;
  const int n0 = tile * 16;
  const __bf16* xcTb = xcT + (size_t)b * HWSZ * CDIM;
  __bf16* st = stage[wave];
  const int n  = lane & 15;
  const int mh = (lane >> 4) << 3;

  v8f q[4] = {};
#pragma unroll
  for (int kb = 0; kb < CDIM; kb += 32) {
    v16bf bf = load_b_nmajor(xcTb, CDIM, n0, kb, lane);
#pragma unroll
    for (int mt = 0; mt < 4; ++mt)
      q[mt] = wmma_bf16(load_a_bf(wq_l, CDIM, mt * 16, kb, lane), bf, q[mt]);
  }
#pragma unroll
  for (int mt = 0; mt < 4; ++mt) {
    Pk8 pk;
#pragma unroll
    for (int r = 0; r < 8; ++r) {
      const int o = mt * 16 + mh + r;
      pk.h[r] = tobf((q[mt][r] + bq[o]) * 0.125f);
    }
    *(uint4*)&st[n * 64 + mt * 16 + mh] = pk.u;
  }
  lds_fence();

  v8f sc[4] = {};
#pragma unroll
  for (int kb = 0; kb < CDIM; kb += 32) {
    v16bf bf = load_b_nmajor(st, 64, 0, kb, lane);
#pragma unroll
    for (int mt = 0; mt < 4; ++mt)
      sc[mt] = wmma_bf16(load_a_bf(K_l, 64, mt * 16, kb, lane), bf, sc[mt]);
  }

  float mx = -1e30f;
#pragma unroll
  for (int mt = 0; mt < 4; ++mt)
#pragma unroll
    for (int r = 0; r < 8; ++r) {
      const int m = mt * 16 + mh + r;
      if (m < NTOK) mx = fmaxf(mx, sc[mt][r]);
    }
  mx = fmaxf(mx, __shfl_xor(mx, 16, 32));
  float sum = 0.f;
#pragma unroll
  for (int mt = 0; mt < 4; ++mt)
#pragma unroll
    for (int r = 0; r < 8; ++r) {
      const int m = mt * 16 + mh + r;
      const float e = (m < NTOK) ? __expf(sc[mt][r] - mx) : 0.f;
      sc[mt][r] = e;
      sum += e;
    }
  sum += __shfl_xor(sum, 16, 32);
  const float inv = 1.f / sum;

#pragma unroll
  for (int mt = 0; mt < 4; ++mt) {
    Pk8 pk;
#pragma unroll
    for (int r = 0; r < 8; ++r) pk.h[r] = tobf(sc[mt][r] * inv);
    *(uint4*)&st[n * 64 + mt * 16 + mh] = pk.u;
  }
  lds_fence();

  v8f xw[4] = {};
#pragma unroll
  for (int kb = 0; kb < 64; kb += 32) {
    v16bf bf = load_b_nmajor(st, 64, 0, kb, lane);
#pragma unroll
    for (int mt = 0; mt < 4; ++mt)
      xw[mt] = wmma_bf16(load_a_bf(V_l, 64, mt * 16, kb, lane), bf, xw[mt]);
  }
#pragma unroll
  for (int mt = 0; mt < 4; ++mt) {
    Pk8 pk;
#pragma unroll
    for (int r = 0; r < 8; ++r) pk.h[r] = tobf(xw[mt][r]);
    *(uint4*)&st[n * 64 + mt * 16 + mh] = pk.u;
  }
  lds_fence();

  float* outb = out + (size_t)b * NC * HWSZ;
#pragma unroll
  for (int mt = 0; mt < 16; ++mt) {
    v8f acc = {};
    acc = wmma_bf16(load_a_bf(wr_l, CDIM, mt * 16, 0, lane),
                    load_b_nmajor(st, 64, 0, 0, lane), acc);
    acc = wmma_bf16(load_a_bf(wr_l, CDIM, mt * 16, 32, lane),
                    load_b_nmajor(st, 64, 0, 32, lane), acc);
#pragma unroll
    for (int r = 0; r < 8; ++r) {
      const int o = mt * 16 + mh + r;
      outb[(size_t)o * HWSZ + n0 + n] = acc[r] + br[o];
    }
  }
}

// ---------------------------------------------------------------------------
extern "C" void kernel_launch(void* const* d_in, const int* in_sizes, int n_in,
                              void* d_out, int out_size, void* d_ws, size_t ws_size,
                              hipStream_t stream) {
  const float* x          = (const float*)d_in[0];
  const float* w_compress = (const float*)d_in[1];
  const float* b_compress = (const float*)d_in[2];
  const float* w_dsc      = (const float*)d_in[3];
  const float* b_dsc      = (const float*)d_in[4];
  const float* ln_g       = (const float*)d_in[5];
  const float* ln_b       = (const float*)d_in[6];
  const float* wq         = (const float*)d_in[7];
  const float* bq         = (const float*)d_in[8];
  const float* wk         = (const float*)d_in[9];
  const float* bk         = (const float*)d_in[10];
  const float* wv         = (const float*)d_in[11];
  const float* bv         = (const float*)d_in[12];
  const float* w_restore  = (const float*)d_in[13];
  const float* b_restore  = (const float*)d_in[14];
  float* out = (float*)d_out;

  char* p = (char*)d_ws;
  float*  xc    = (float*)p;   p += (size_t)NB * CDIM * HWSZ * 4;
  __bf16* xcT   = (__bf16*)p;  p += (size_t)NB * HWSZ * CDIM * 2;
  __bf16* Kp    = (__bf16*)p;  p += (size_t)NB * 64 * 64 * 2;
  __bf16* Vt    = (__bf16*)p;  p += (size_t)NB * 64 * 64 * 2;
  __bf16* wc_bf = (__bf16*)p;  p += (size_t)CDIM * NC * 2;
  __bf16* wq_bf = (__bf16*)p;  p += (size_t)CDIM * CDIM * 2;
  __bf16* wr_bf = (__bf16*)p;  p += (size_t)NC * CDIM * 2;

  convert_weights_kernel<<<64, 256, 0, stream>>>(w_compress, wq, w_restore,
                                                 wc_bf, wq_bf, wr_bf);
  compress_kernel<<<dim3((NTILE + WPB - 1) / WPB, NB), TPB, 0, stream>>>(
      x, wc_bf, b_compress, xc, xcT);
  pool_kv_kernel<<<dim3(NB), 256, 0, stream>>>(xc, w_dsc, b_dsc, ln_g, ln_b,
                                               wk, bk, wv, bv, Kp, Vt);
  attn_restore_kernel<<<dim3((NTILE + WPB - 1) / WPB, NB), TPB, 0, stream>>>(
      xcT, wq_bf, bq, Kp, Vt, wr_bf, b_restore, out);
}
